// MAB_2267742733048
// MI455X (gfx1250) — compile-verified, matrix operand order checked by reading, not run on hardware
//
#include <hip/hip_runtime.h>
#include <hip/hip_bf16.h>

// ---------------------------------------------------------------------------
// MAB (multi-head attention block) for MI455X / gfx1250, wave32, bf16 WMMA,
// with async global->LDS double-buffered staging.
// B=8, NQ=NK=1024, QDIM=KDIM=VDIM=512, H=8, DS=64.
// ---------------------------------------------------------------------------

typedef __attribute__((ext_vector_type(16))) __bf16 v16bf;
typedef __attribute__((ext_vector_type(8)))  float  v8f;
typedef __attribute__((ext_vector_type(8)))  unsigned int v8u;
typedef __attribute__((ext_vector_type(4)))  int    v4i_t;

// ---- async global->LDS support probe --------------------------------------
#if defined(__has_builtin)
#  if __has_builtin(__builtin_amdgcn_global_load_async_to_lds_b128) && \
      __has_builtin(__builtin_amdgcn_s_wait_asynccnt)
#    define HAVE_ASYNC_COPY 1
#  endif
#endif
#ifndef HAVE_ASYNC_COPY
#  define HAVE_ASYNC_COPY 0
#  warning "gfx1250 async-to-LDS builtins not found; falling back to synchronous staging"
#endif

#if HAVE_ASYNC_COPY
typedef __attribute__((address_space(1))) v4i_t gv4i_t;   // global 16B chunk
typedef __attribute__((address_space(3))) v4i_t lv4i_t;   // LDS    16B chunk
#endif

// copy one 16-byte chunk global -> LDS (per-lane addresses)
__device__ __forceinline__ void cp16B(void* dst_lds, const void* src_glob) {
#if HAVE_ASYNC_COPY
    __builtin_amdgcn_global_load_async_to_lds_b128(
        (gv4i_t*)(unsigned long long)src_glob,
        (lv4i_t*)(unsigned int)(unsigned long long)dst_lds,
        0, 0);
#else
    *(uint4*)dst_lds = *(const uint4*)src_glob;
#endif
}
__device__ __forceinline__ void wait_async() {
#if HAVE_ASYNC_COPY
    __builtin_amdgcn_s_wait_asynccnt(0);
#endif
}

// ---- scalar helpers --------------------------------------------------------
__device__ __forceinline__ unsigned short f2bf(float f) {
    unsigned int u = __builtin_bit_cast(unsigned int, f);
    u += 0x7fffu + ((u >> 16) & 1u);          // round-to-nearest-even
    return (unsigned short)(u >> 16);
}
__device__ __forceinline__ float bf2f(unsigned short s) {
    unsigned int u = ((unsigned int)s) << 16;
    return __builtin_bit_cast(float, u);
}

__device__ __forceinline__ v8f wmma_bf16(v16bf a, v16bf b, v8f c) {
    return __builtin_amdgcn_wmma_f32_16x16x32_bf16(
        false, a, false, b, (short)0, c, false, false);
}

// 16x32 bf16 WMMA fragment from LDS (A layout; B tiles are stored
// column-major so lane n supplies column n with the same loader).
__device__ __forceinline__ v16bf load_frag(const unsigned short* lds,
                                           int row0, int stride, int koff) {
    int lane = threadIdx.x & 31;
    const unsigned short* p = lds + (row0 + (lane & 15)) * stride + koff
                                  + ((lane & 16) ? 8 : 0);
    uint4 lo = *(const uint4*)(p);
    uint4 hi = *(const uint4*)(p + 16);
    v8u u;
    u[0] = lo.x; u[1] = lo.y; u[2] = lo.z; u[3] = lo.w;
    u[4] = hi.x; u[5] = hi.y; u[6] = hi.z; u[7] = hi.w;
    return __builtin_bit_cast(v16bf, u);
}

__device__ __forceinline__ v8f v8f_zero() {
    v8f z;
    #pragma unroll
    for (int e = 0; e < 8; ++e) z[e] = 0.0f;
    return z;
}

// ---------------------------------------------------------------------------
// Prep: f32 -> bf16 flat convert (X, Y).  n must be a multiple of 1024.
// ---------------------------------------------------------------------------
__global__ __launch_bounds__(256)
void cvt_kernel(const float* __restrict__ in, unsigned short* __restrict__ out) {
    size_t i = ((size_t)blockIdx.x * 256 + threadIdx.x) * 4;
    float4 v = *(const float4*)(in + i);
    unsigned int lo = (unsigned int)f2bf(v.x) | ((unsigned int)f2bf(v.y) << 16);
    unsigned int hi = (unsigned int)f2bf(v.z) | ((unsigned int)f2bf(v.w) << 16);
    uint2 p; p.x = lo; p.y = hi;
    *(uint2*)(out + i) = p;
}

// Prep: W[512,512] f32 -> Wt[n][k] bf16 (transpose + convert), 32x32 tiles.
__global__ __launch_bounds__(256)
void wt_kernel(const float* __restrict__ W, unsigned short* __restrict__ Wt) {
    __shared__ float t[32][33];
    const int lx = threadIdx.x & 31, ly = threadIdx.x >> 5;   // 32 x 8
    const int kb = blockIdx.x * 32, nb = blockIdx.y * 32;
    #pragma unroll
    for (int rr = 0; rr < 32; rr += 8)
        t[ly + rr][lx] = W[(size_t)(kb + ly + rr) * 512 + nb + lx];
    __syncthreads();
    #pragma unroll
    for (int rr = 0; rr < 32; rr += 8)
        Wt[(size_t)(nb + ly + rr) * 512 + kb + lx] = f2bf(t[lx][ly + rr]);
}

// ---------------------------------------------------------------------------
// GEMM: C[M=8192,N=512] = A[M,512] x W[512,512] + bias (A bf16, Wt bf16 [n][k])
// 128x128 tile / 256 threads, k-step 32, double-buffered async LDS staging.
// MODE 0: bf16 row-major   MODE 1: f32 row-major
// MODE 2: bf16 per-head transposed  Vt[((b*8+h)*64+d)*1024 + nk]
// ---------------------------------------------------------------------------
template<int MODE>
__global__ __launch_bounds__(256)
void gemm_kernel(const unsigned short* __restrict__ A,
                 const unsigned short* __restrict__ Bt,
                 const float* __restrict__ bias, void* __restrict__ outp) {
    __shared__ __align__(16) unsigned short As[2][128][40];  // [m][k]
    __shared__ __align__(16) unsigned short Bs[2][128][40];  // [n][k]

    const int tid  = threadIdx.x;
    const int wave = tid >> 5;
    const int lane = tid & 31;
    const int m0 = blockIdx.y * 128;
    const int n0 = blockIdx.x * 128;
    const int wm = (wave >> 1) * 32;
    const int wn = (wave & 1) * 64;

    // stage one 128x32 bf16 tile pair (A and Bt); 512 16B-chunks each
    auto stage = [&](int buf, int k0) {
        #pragma unroll
        for (int t = 0; t < 2; ++t) {
            int c = tid + t * 256;              // 0..511
            int r = c >> 2, kc = (c & 3) * 8;   // row, k-chunk (shorts)
            cp16B(&As[buf][r][kc], &A [(size_t)(m0 + r) * 512 + k0 + kc]);
            cp16B(&Bs[buf][r][kc], &Bt[(size_t)(n0 + r) * 512 + k0 + kc]);
        }
    };

    v8f acc[2][4];
    #pragma unroll
    for (int i = 0; i < 2; ++i)
        #pragma unroll
        for (int j = 0; j < 4; ++j) acc[i][j] = v8f_zero();

    stage(0, 0);
    for (int ks = 0; ks < 16; ++ks) {
        wait_async();
        __syncthreads();                        // tile ks fully in LDS
        if (ks + 1 < 16) stage((ks + 1) & 1, (ks + 1) * 32);

        const unsigned short* Ab = &As[ks & 1][0][0];
        const unsigned short* Bb = &Bs[ks & 1][0][0];
        v16bf a0 = load_frag(Ab, wm,      40, 0);
        v16bf a1 = load_frag(Ab, wm + 16, 40, 0);
        #pragma unroll
        for (int j = 0; j < 4; ++j) {
            v16bf bf = load_frag(Bb, wn + j * 16, 40, 0);
            acc[0][j] = wmma_bf16(a0, bf, acc[0][j]);
            acc[1][j] = wmma_bf16(a1, bf, acc[1][j]);
        }
    }

    #pragma unroll
    for (int i = 0; i < 2; ++i) {
        #pragma unroll
        for (int j = 0; j < 4; ++j) {
            int Mb = m0 + wm + i * 16 + ((lane & 16) ? 8 : 0);
            int Nb = n0 + wn + j * 16 + (lane & 15);
            float bia = bias[Nb];
            #pragma unroll
            for (int r = 0; r < 8; ++r) {
                int Mi = Mb + r;
                float v = acc[i][j][r] + bia;
                if (MODE == 0) {
                    ((unsigned short*)outp)[(size_t)Mi * 512 + Nb] = f2bf(v);
                } else if (MODE == 1) {
                    ((float*)outp)[(size_t)Mi * 512 + Nb] = v;
                } else {
                    int b = Mi >> 10, nk = Mi & 1023;
                    int h = Nb >> 6,  d  = Nb & 63;
                    ((unsigned short*)outp)[(((size_t)(b * 8 + h) * 64 + d) << 10) + nk] = f2bf(v);
                }
            }
        }
    }
}

// ---------------------------------------------------------------------------
// Flash attention + residual; double-buffered async K/V staging.
// ---------------------------------------------------------------------------
constexpr size_t OFF_QS = 0;                           // Qs [128][72] u16
constexpr size_t OFF_KS = OFF_QS + 128 * 72 * 2;       // Ks 2x [128][72] u16
constexpr size_t OFF_VS = OFF_KS + 2 * 128 * 72 * 2;   // Vs 2x [64][136] u16
constexpr size_t OFF_PS = OFF_VS + 2 * 64 * 136 * 2;   // Ps [128][136] u16
constexpr size_t OFF_SS = OFF_PS + 128 * 136 * 2;      // Ss [128][132] f32
constexpr size_t OFF_RM = OFF_SS + 128 * 132 * 4;
constexpr size_t OFF_RL = OFF_RM + 128 * 4;
constexpr size_t OFF_RF = OFF_RL + 128 * 4;
constexpr size_t ATTN_LDS_BYTES = OFF_RF + 128 * 4;    // 194048 B < 320 KB

__global__ __launch_bounds__(256)
void attn_kernel(const unsigned short* __restrict__ Qbf,
                 const unsigned short* __restrict__ Kbf,
                 const unsigned short* __restrict__ Vt,
                 float* __restrict__ O) {
    extern __shared__ __align__(16) char smem[];
    unsigned short* Qs = (unsigned short*)(smem + OFF_QS);
    unsigned short* Ks = (unsigned short*)(smem + OFF_KS);
    unsigned short* Vs = (unsigned short*)(smem + OFF_VS);
    unsigned short* Ps = (unsigned short*)(smem + OFF_PS);
    float* Ss   = (float*)(smem + OFF_SS);
    float* rowm = (float*)(smem + OFF_RM);
    float* rowl = (float*)(smem + OFF_RL);
    float* rowf = (float*)(smem + OFF_RF);

    const int tid  = threadIdx.x;
    const int wave = tid >> 5;
    const int lane = tid & 31;
    const int q0 = blockIdx.x * 128;
    const int h  = blockIdx.y;
    const int b  = blockIdx.z;

    const size_t qbase = ((size_t)b * 1024 + q0) * 512 + h * 64;
    const size_t kbase = (size_t)b * 1024 * 512 + h * 64;
    const size_t vbase = ((size_t)(b * 8 + h) * 64) * 1024;

    // stage Q tile (1024 16B chunks)
    #pragma unroll
    for (int t = 0; t < 4; ++t) {
        int c = tid + t * 256;
        int r = c >> 3, kc = (c & 7) * 8;
        cp16B(&Qs[r * 72 + kc], &Qbf[qbase + (size_t)r * 512 + kc]);
    }
    // stage one K/V tile pair for k-block k0 into buffer `buf`
    auto stageKV = [&](int buf, int k0) {
        unsigned short* Kb = Ks + buf * (128 * 72);
        unsigned short* Vb = Vs + buf * (64 * 136);
        #pragma unroll
        for (int t = 0; t < 4; ++t) {           // K: 128 rows x 8 chunks
            int c = tid + t * 256;
            int r = c >> 3, kc = (c & 7) * 8;
            cp16B(&Kb[r * 72 + kc], &Kbf[kbase + (size_t)(k0 + r) * 512 + kc]);
        }
        #pragma unroll
        for (int t = 0; t < 4; ++t) {           // V: 64 d-rows x 16 chunks
            int c = tid + t * 256;
            int d = c >> 4, kc = (c & 15) * 8;
            cp16B(&Vb[d * 136 + kc], &Vt[vbase + (size_t)d * 1024 + k0 + kc]);
        }
    };

    if (tid < 128) { rowm[tid] = -1e30f; rowl[tid] = 0.0f; }

    v8f oacc[4];
    #pragma unroll
    for (int j = 0; j < 4; ++j) oacc[j] = v8f_zero();

    stageKV(0, 0);
    for (int kt = 0; kt < 8; ++kt) {
        wait_async();
        __syncthreads();                         // Q + K/V tile kt resident
        if (kt + 1 < 8) stageKV((kt + 1) & 1, (kt + 1) * 128);

        const unsigned short* Kb = Ks + (kt & 1) * (128 * 72);
        const unsigned short* Vb = Vs + (kt & 1) * (64 * 136);

        // S = Q K^T / 8 ; wave owns a 16-row strip, 128 cols
        v8f sacc[8];
        #pragma unroll
        for (int j = 0; j < 8; ++j) sacc[j] = v8f_zero();
        #pragma unroll
        for (int ko = 0; ko < 64; ko += 32) {
            v16bf a = load_frag(Qs, wave * 16, 72, ko);
            #pragma unroll
            for (int j = 0; j < 8; ++j) {
                v16bf bfrag = load_frag(Kb, j * 16, 72, ko);
                sacc[j] = wmma_bf16(a, bfrag, sacc[j]);
            }
        }
        #pragma unroll
        for (int j = 0; j < 8; ++j)
            #pragma unroll
            for (int r = 0; r < 8; ++r) {
                int M = wave * 16 + ((lane & 16) ? 8 : 0) + r;
                int N = j * 16 + (lane & 15);
                Ss[M * 132 + N] = sacc[j][r] * 0.125f;
            }
        __syncthreads();

        // online softmax: one thread per row
        if (tid < 128) {
            int r = tid;
            float mloc = -1e30f;
            for (int c = 0; c < 128; ++c) mloc = fmaxf(mloc, Ss[r * 132 + c]);
            float mold = rowm[r];
            float mnew = fmaxf(mold, mloc);
            float fac  = __expf(mold - mnew);
            float s = 0.0f;
            for (int c = 0; c < 128; ++c) {
                float p = __expf(Ss[r * 132 + c] - mnew);
                Ps[r * 136 + c] = f2bf(p);
                s += p;
            }
            rowl[r] = rowl[r] * fac + s;
            rowm[r] = mnew;
            rowf[r] = fac;
        }
        __syncthreads();

        // rescale running O, then O += P * V
        #pragma unroll
        for (int r = 0; r < 8; ++r) {
            int M = wave * 16 + ((lane & 16) ? 8 : 0) + r;
            float fac = rowf[M];
            #pragma unroll
            for (int j = 0; j < 4; ++j) oacc[j][r] *= fac;
        }
        #pragma unroll
        for (int ko = 0; ko < 128; ko += 32) {
            v16bf a = load_frag(Ps, wave * 16, 136, ko);
            #pragma unroll
            for (int j = 0; j < 4; ++j) {
                v16bf bfrag = load_frag(Vb, j * 16, 136, ko);
                oacc[j] = wmma_bf16(a, bfrag, oacc[j]);
            }
        }
    }

    // O = Q + softmax(S) V   (f32 [B,NQ,512], head column band)
    #pragma unroll
    for (int r = 0; r < 8; ++r) {
        int Mloc = wave * 16 + ((lane & 16) ? 8 : 0) + r;
        float inv = 1.0f / rowl[Mloc];
        #pragma unroll
        for (int j = 0; j < 4; ++j) {
            int d = j * 16 + (lane & 15);
            float q = bf2f(Qs[Mloc * 72 + d]);
            O[((size_t)b * 1024 + q0 + Mloc) * 512 + h * 64 + d] = q + oacc[j][r] * inv;
        }
    }
}

// ---------------------------------------------------------------------------
// Row LayerNorm over 512 elements. One 256-thread block per row.
// ---------------------------------------------------------------------------
template<bool OUT_BF16>
__global__ __launch_bounds__(256)
void ln_kernel(const float* __restrict__ X, const float* __restrict__ g,
               const float* __restrict__ bta, void* __restrict__ out) {
    __shared__ float red[256];
    const int row = blockIdx.x;
    const float* x = X + (size_t)row * 512;

    float s = 0.0f;
    for (int i = threadIdx.x; i < 512; i += 256) s += x[i];
    red[threadIdx.x] = s; __syncthreads();
    for (int off = 128; off > 0; off >>= 1) {
        if (threadIdx.x < off) red[threadIdx.x] += red[threadIdx.x + off];
        __syncthreads();
    }
    float mu = red[0] * (1.0f / 512.0f);
    __syncthreads();

    float v = 0.0f;
    for (int i = threadIdx.x; i < 512; i += 256) { float d = x[i] - mu; v += d * d; }
    red[threadIdx.x] = v; __syncthreads();
    for (int off = 128; off > 0; off >>= 1) {
        if (threadIdx.x < off) red[threadIdx.x] += red[threadIdx.x + off];
        __syncthreads();
    }
    float rstd = rsqrtf(red[0] * (1.0f / 512.0f) + 1e-5f);

    for (int i = threadIdx.x; i < 512; i += 256) {
        float y = (x[i] - mu) * rstd * g[i] + bta[i];
        if (OUT_BF16) ((unsigned short*)out)[(size_t)row * 512 + i] = f2bf(y);
        else          ((float*)out)[(size_t)row * 512 + i] = y;
    }
}

// ---------------------------------------------------------------------------
extern "C" void kernel_launch(void* const* d_in, const int* in_sizes, int n_in,
                              void* d_out, int out_size, void* d_ws, size_t ws_size,
                              hipStream_t stream) {
    const float* X  = (const float*)d_in[0];
    const float* Y  = (const float*)d_in[1];
    const float* Wq = (const float*)d_in[2];
    const float* bq = (const float*)d_in[3];
    const float* Wk = (const float*)d_in[4];
    const float* bk = (const float*)d_in[5];
    const float* Wv = (const float*)d_in[6];
    const float* bv = (const float*)d_in[7];
    const float* Wo = (const float*)d_in[8];
    const float* bo = (const float*)d_in[9];
    const float* g1 = (const float*)d_in[10];
    const float* b1 = (const float*)d_in[11];
    const float* g2 = (const float*)d_in[12];
    const float* b2 = (const float*)d_in[13];
    float* out = (float*)d_out;

    char* ws = (char*)d_ws;
    const size_t MB = 1u << 20;
    unsigned short* Xbf = (unsigned short*)(ws);             //  8 MB
    unsigned short* Ybf = (unsigned short*)(ws +  8 * MB);   //  8 MB
    unsigned short* Wqt = (unsigned short*)(ws + 16 * MB);   // 512 KB each
    unsigned short* Wkt = Wqt + 512 * 512;
    unsigned short* Wvt = Wkt + 512 * 512;
    unsigned short* Wot = Wvt + 512 * 512;
    unsigned short* Qbf = (unsigned short*)(ws + 18 * MB);   //  8 MB
    unsigned short* Kbf = (unsigned short*)(ws + 26 * MB);   //  8 MB
    unsigned short* Vt  = (unsigned short*)(ws + 34 * MB);   //  8 MB
    float*          O1  = (float*)         (ws + 42 * MB);   // 16 MB (reused as O2)
    unsigned short* N1  = (unsigned short*)(ws + 58 * MB);   //  8 MB  -> 66 MB total

    // prep: bf16 activations + transposed bf16 weights
    cvt_kernel<<<4096, 256, 0, stream>>>(X, Xbf);
    cvt_kernel<<<4096, 256, 0, stream>>>(Y, Ybf);
    dim3 wtg(16, 16);
    wt_kernel<<<wtg, 256, 0, stream>>>(Wq, Wqt);
    wt_kernel<<<wtg, 256, 0, stream>>>(Wk, Wkt);
    wt_kernel<<<wtg, 256, 0, stream>>>(Wv, Wvt);
    wt_kernel<<<wtg, 256, 0, stream>>>(Wo, Wot);

    dim3 gblk(4, 64);   // N tiles x M tiles
    gemm_kernel<0><<<gblk, 256, 0, stream>>>(Xbf, Wqt, bq, Qbf);
    gemm_kernel<0><<<gblk, 256, 0, stream>>>(Ybf, Wkt, bk, Kbf);
    gemm_kernel<2><<<gblk, 256, 0, stream>>>(Ybf, Wvt, bv, Vt);

    (void)hipFuncSetAttribute((const void*)attn_kernel,
                              hipFuncAttributeMaxDynamicSharedMemorySize,
                              (int)ATTN_LDS_BYTES);
    attn_kernel<<<dim3(8, 8, 8), 256, ATTN_LDS_BYTES, stream>>>(Qbf, Kbf, Vt, O1);

    ln_kernel<true ><<<8192, 256, 0, stream>>>(O1, g1, b1, N1);
    gemm_kernel<1><<<gblk, 256, 0, stream>>>(N1, Wot, bo, O1);
    ln_kernel<false><<<8192, 256, 0, stream>>>(O1, g2, b2, out);
}
